// NeuralODEModel_78649441124434
// MI455X (gfx1250) — matrix operand order repaired and checked by reading, not compiled, not run on hardware
//
#include <hip/hip_runtime.h>
#include <hip/hip_bf16.h>

typedef __attribute__((ext_vector_type(16))) _Float16 v16h;
typedef __attribute__((ext_vector_type(8)))  float    v8f;

#define AUG   8
#define LAT   32
#define NOUT  (1 + AUG + LAT)   // 41 state dims
#define NIN   (NOUT + 2)        // 43 MLP inputs
#define HID   128
#define ROWS  16                // batch rows per workgroup (one WMMA M tile)
#define XP    72                // sX pitch (halves), even, >= 64
#define HP    136               // hidden pitch (halves), even, >= 128
#define YP    48                // state pitch (floats)

static __device__ __forceinline__ v8f wmma_f16(v16h a, v16h b, v8f c) {
  // D = A(16x32 f16) * B(32x16 f16) + C(16x16 f32)
  return __builtin_amdgcn_wmma_f32_16x16x32_f16(false, a, false, b, (short)0, c,
                                                false, false);
}

// Branch-free tanh: CDNA5 V_TANH_F32 (confirmed lowering); exp/rcp fallback.
static __device__ __forceinline__ float fast_tanh(float x) {
#if __has_builtin(__builtin_amdgcn_tanhf)
  return __builtin_amdgcn_tanhf(x);
#else
  float e = __expf(-2.0f * fabsf(x));
  float t = (1.0f - e) * __builtin_amdgcn_rcpf(1.0f + e);
  return copysignf(t, x);
#endif
}

// A fragment (16x32 f16) gathered from LDS row-major [16][pitch].
// ISA layout: lane L holds row m=L%16; VGPR e holds K pair
//   kb = (e/4)*16 + (L/16)*8 + 2*(e%4)   (contiguous -> ds_load_b128 pairs)
static __device__ __forceinline__ v16h load_afrag(const _Float16* base, int pitch,
                                                  int m, int hi, int kofs) {
  const uint32_t* w = (const uint32_t*)base;
  union { v16h v; uint32_t u[8]; } r;
#pragma unroll
  for (int e = 0; e < 8; ++e) {
    int kb = ((e >> 2) << 4) + (hi << 3) + ((e & 3) << 1) + kofs;
    r.u[e] = w[(m * pitch + kb) >> 1];
  }
  return r.v;
}

// B fragment (32x16 f16) built once from global f32 weights W[n*ld + k].
// Lane L holds column n=L%16; VGPR e holds K pair kb = (e/4)*16 + (L/16)*8 + 2*(e%4).
static __device__ __forceinline__ v16h make_bfrag(const float* W, int n, int ld,
                                                  int kofs, int kmax, int hi) {
  union { v16h v; _Float16 h[16]; } r;
#pragma unroll
  for (int e = 0; e < 8; ++e) {
    int kb = ((e >> 2) << 4) + (hi << 3) + ((e & 3) << 1) + kofs;
    float v0 = (kb     < kmax) ? W[n * ld + kb]     : 0.0f;
    float v1 = (kb + 1 < kmax) ? W[n * ld + kb + 1] : 0.0f;
    r.h[2 * e]     = (_Float16)v0;
    r.h[2 * e + 1] = (_Float16)v1;
  }
  return r.v;
}

__global__ __launch_bounds__(256) void node_rk4_wmma_kernel(
    const float* __restrict__ ts, const float* __restrict__ y0,
    const float* __restrict__ latent, const int* __restrict__ length,
    const float* __restrict__ dts_g, const float* __restrict__ dcs,
    const float* __restrict__ W1, const float* __restrict__ b1,
    const float* __restrict__ W2, const float* __restrict__ b2,
    const float* __restrict__ W3, const float* __restrict__ b3,
    float* __restrict__ out, int B, int T, int D) {
  __shared__ _Float16 sX [ROWS * XP];   // f16 MLP input  [16][72]
  __shared__ _Float16 sH1[ROWS * HP];   // f16 hidden 1   [16][136]
  __shared__ _Float16 sH2[ROWS * HP];   // f16 hidden 2   [16][136]
  __shared__ float    sY [ROWS * YP];   // f32 state      [16][48]
  __shared__ float    sAcc[ROWS * 16];  // RK4 accumulator (live 16 cols)
  __shared__ float    sK [ROWS * 16];   // vf output k    (live 16 cols)
  __shared__ float    sTend[ROWS];
  __shared__ float    sDts[1024];       // dense time grid (D<=1024 in ref: 256)

  const int tid   = threadIdx.x;
  const int wave  = tid >> 5;
  const int lane  = tid & 31;
  const int hi    = lane >> 4;
  const int mlane = lane & 15;          // A row / B-C column within the tile
  const int wg    = blockIdx.x;
  const int Ds    = (D <= 1024) ? D : 1024;

  // ---- build invariant weight fragments in registers (once) ----
  const int ncol = wave * 16 + mlane;   // hidden column owned for layers 1/2
  v16h w1f[2], w2f[4], w3f[4];
#pragma unroll
  for (int k = 0; k < 2; ++k) w1f[k] = make_bfrag(W1, ncol, NIN, 32 * k, NIN, hi);
#pragma unroll
  for (int k = 0; k < 4; ++k) w2f[k] = make_bfrag(W2, ncol, HID, 32 * k, HID, hi);
#pragma unroll
  for (int k = 0; k < 4; ++k) w3f[k] = make_bfrag(W3, mlane, HID, 32 * k, HID, hi);
  const float bias1 = b1[ncol];
  const float bias2 = b2[ncol];
  const float bias3 = b3[mlane < NOUT ? mlane : 0];

  // ---- per-block init ----
  for (int idx = tid; idx < ROWS * XP; idx += 256) sX[idx] = (_Float16)0.0f;
  for (int idx = tid; idx < Ds; idx += 256) sDts[idx] = dts_g[idx];
  __syncthreads();

  if (tid < ROWS) {
    int m   = tid;
    int row = wg * ROWS + m;
    int rc  = row < B ? row : B - 1;
    float y0v = y0[rc];
    sY[m * YP + 0] = y0v;
#pragma unroll
    for (int j = 1; j <= AUG; ++j) sY[m * YP + j] = 0.0f;
    for (int j = 0; j < LAT; ++j) sY[m * YP + 1 + AUG + j] = latent[rc * LAT + j];
    for (int j = NOUT; j < YP; ++j) sY[m * YP + j] = 0.0f;
    // constant latent part of the MLP input (cols 16..40), written once
    for (int j = 16; j < NOUT; ++j) sX[m * XP + j] = (_Float16)sY[m * YP + j];
    int len = length[rc];
    int li  = len - 1; if (li < 0) li = 0;
    sTend[m] = ts[li];
    if (row < B) out[row * T + 0] = y0v;
  }
  __syncthreads();

  // per-row forcing interpolation + time column (threads 0..15).
  // Only called when t actually changes: k1's t always equals the previous
  // eval's t (seeded once before the loop), and k3 shares k2's t.
  auto fill_tc = [&](float t) {
    if (tid < ROWS) {
      int m = tid;
      sX[m * XP + NOUT] = (_Float16)t;          // col 41 = t
      int lo = 0, hb = Ds;
      while (lo < hb) { int mid = (lo + hb) >> 1;
                        if (sDts[mid] < t) lo = mid + 1; else hb = mid; }
      int i1 = lo; if (i1 < 1) i1 = 1; if (i1 > Ds - 1) i1 = Ds - 1;
      float tl = sDts[i1 - 1], tr = sDts[i1];
      float w = (t - tl) * __builtin_amdgcn_rcpf(tr - tl);
      w = fminf(fmaxf(w, 0.0f), 1.0f);
      int row = wg * ROWS + m;
      int rc  = row < B ? row : B - 1;
      float cv = (1.0f - w) * dcs[rc * D + i1 - 1] + w * dcs[rc * D + i1];
      sX[m * XP + NOUT + 1] = (_Float16)cv;     // col 42 = c
    }
  };

  // one vector-field evaluation: sX -> sK (live 16 output cols).
  // All A fragments of a layer are gathered before the WMMA chain so the
  // ds_load_b128s issue back-to-back and overlap the matrix pipe.
  auto mlp = [&](float tcur) {
    // layer 1: K = 64 (padded 43)
    v8f c1 = {};
    {
      v16h a[2];
      a[0] = load_afrag(sX, XP, mlane, hi, 0);
      a[1] = load_afrag(sX, XP, mlane, hi, 32);
      c1 = wmma_f16(a[0], w1f[0], c1);
      c1 = wmma_f16(a[1], w1f[1], c1);
    }
#pragma unroll
    for (int v = 0; v < 8; ++v) {
      int r = v + (hi << 3);
      sH1[r * HP + ncol] = (_Float16)fast_tanh(c1[v] + bias1);
    }
    __syncthreads();
    // layer 2: K = 128
    v8f c2 = {};
    {
      v16h a[4];
#pragma unroll
      for (int k = 0; k < 4; ++k) a[k] = load_afrag(sH1, HP, mlane, hi, 32 * k);
#pragma unroll
      for (int k = 0; k < 4; ++k) c2 = wmma_f16(a[k], w2f[k], c2);
    }
#pragma unroll
    for (int v = 0; v < 8; ++v) {
      int r = v + (hi << 3);
      sH2[r * HP + ncol] = (_Float16)fast_tanh(c2[v] + bias2);
    }
    __syncthreads();
    // layer 3: only cols 0..15 are live (latent dims are frozen) -> wave 0
    if (wave == 0) {
      v8f c3 = {};
      v16h a[4];
#pragma unroll
      for (int k = 0; k < 4; ++k) a[k] = load_afrag(sH2, HP, mlane, hi, 32 * k);
#pragma unroll
      for (int k = 0; k < 4; ++k) c3 = wmma_f16(a[k], w3f[k], c3);
#pragma unroll
      for (int v = 0; v < 8; ++v) {
        int r = v + (hi << 3);
        float o = c3[v] + bias3;
        if (mlane == 0) o = -__cosf(o);         // constrained dim
        if (mlane >= 1 + AUG) o = 0.0f;         // frozen latent dims
        if (tcur > sTend[r]) o = 0.0f;          // per-sample stop time
        sK[r * 16 + mlane] = o;
      }
    }
    __syncthreads();
  };

  const int m16 = tid >> 4, c16 = tid & 15;

  // seed t/c columns for the very first evaluation
  fill_tc(ts[0]);

  // ---- time integration: 2 RK4 substeps per save interval ----
#pragma unroll 1
  for (int i = 0; i + 1 < T; ++i) {
    float ta = ts[i], tb = ts[i + 1];
    float dt = (tb - ta) * 0.5f;                // SUB = 2
#pragma unroll 1
    for (int s = 0; s < 2; ++s) {
      float t0 = ta + dt * (float)s;
      float th = 0.5f * dt;
      // k1 setup: x = Y, acc = Y  (t/c columns already hold t0)
      {
        float yv = sY[m16 * YP + c16];
        sAcc[tid] = yv;
        sX[m16 * XP + c16] = (_Float16)yv;
      }
      __syncthreads();
      mlp(t0);                                  // k1
      {
        float kv = sK[tid];
        sAcc[tid] += (dt * (1.0f / 6.0f)) * kv;
        sX[m16 * XP + c16] = (_Float16)(sY[m16 * YP + c16] + th * kv);
        fill_tc(t0 + th);
      }
      __syncthreads();
      mlp(t0 + th);                             // k2
      {
        float kv = sK[tid];
        sAcc[tid] += (dt * (1.0f / 3.0f)) * kv;
        sX[m16 * XP + c16] = (_Float16)(sY[m16 * YP + c16] + th * kv);
        // t unchanged (k3 evaluates at the same midpoint): no fill_tc
      }
      __syncthreads();
      mlp(t0 + th);                             // k3
      {
        float kv = sK[tid];
        sAcc[tid] += (dt * (1.0f / 3.0f)) * kv;
        sX[m16 * XP + c16] = (_Float16)(sY[m16 * YP + c16] + dt * kv);
        fill_tc(t0 + dt);
      }
      __syncthreads();
      mlp(t0 + dt);                             // k4
      {
        float kv = sK[tid];
        sY[m16 * YP + c16] = sAcc[tid] + (dt * (1.0f / 6.0f)) * kv;
      }
      __syncthreads();
    }
    if (tid < ROWS) {
      int row = wg * ROWS + tid;
      if (row < B) out[row * T + (i + 1)] = sY[tid * YP + 0];
    }
  }
}

extern "C" void kernel_launch(void* const* d_in, const int* in_sizes, int n_in,
                              void* d_out, int out_size, void* d_ws, size_t ws_size,
                              hipStream_t stream) {
  const float* ts       = (const float*)d_in[0];
  const float* y0       = (const float*)d_in[1];
  const float* latent   = (const float*)d_in[2];
  const int*   length   = (const int*)d_in[3];
  const float* dense_ts = (const float*)d_in[4];
  const float* dense_cs = (const float*)d_in[5];
  const float* W1 = (const float*)d_in[6];
  const float* b1 = (const float*)d_in[7];
  const float* W2 = (const float*)d_in[8];
  const float* b2 = (const float*)d_in[9];
  const float* W3 = (const float*)d_in[10];
  const float* b3 = (const float*)d_in[11];
  int T = in_sizes[0];
  int B = in_sizes[1];
  int D = in_sizes[4];
  int grid = (B + ROWS - 1) / ROWS;
  node_rk4_wmma_kernel<<<grid, 256, 0, stream>>>(
      ts, y0, latent, length, dense_ts, dense_cs,
      W1, b1, W2, b2, W3, b3, (float*)d_out, B, T, D);
}
